// TemporalAttention3_55138790146544
// MI455X (gfx1250) — matrix operand order repaired and checked by reading, not compiled, not run on hardware
//
#include <hip/hip_runtime.h>

#define T_SEQ   2048
#define B_BATCH 8
#define DFEAT   512
#define GATE3   1536
#define WIN     12
#define NSEQ    (B_BATCH * T_SEQ)

#define MT      2            // WMMA M-tiles (sequence tiles) per block
#define MSEQ    (16 * MT)    // 32 sequences per block
#define NWAVES  8
#define DSLICE  64           // hidden dims per wave
#define NSUB    4            // 16-wide column subtiles per wave
#define KT      (DFEAT / 32) // 16 k-steps per GEMM

#define SECT_Z  (512u  * DFEAT * 2u)   // byte offset of z-gate rows (524288)
#define SECT_N  (1024u * DFEAT * 2u)   // byte offset of n-gate rows (1048576)

typedef __attribute__((ext_vector_type(16))) __bf16        bf16x16;
typedef __attribute__((ext_vector_type(8)))  float         f32x8;
typedef __attribute__((ext_vector_type(4)))  unsigned int  u32x4;

__device__ __forceinline__ unsigned short f2bf_bits(float f) {
  unsigned int u = __builtin_bit_cast(unsigned int, f);
  u += 0x7FFFu + ((u >> 16) & 1u);   // round-to-nearest-even
  return (unsigned short)(u >> 16);
}

__device__ __forceinline__ f32x8 wmma_bf16(bf16x16 a, bf16x16 b, f32x8 c) {
  return __builtin_amdgcn_wmma_f32_16x16x32_bf16(false, a, false, b, (short)0, c,
                                                 false, false);
}

// fast pointwise ops on the v_exp/v_rcp transcendental pipe
__device__ __forceinline__ float fast_sigmoid(float v) {
  return __builtin_amdgcn_rcpf(1.f + __expf(-v));
}
__device__ __forceinline__ float fast_tanh(float v) {
  // tanh(x) = 1 - 2/(exp(2x)+1)
  return 1.f - 2.f * __builtin_amdgcn_rcpf(__expf(2.f * v) + 1.f);
}

// ---- CDNA5 async global->LDS copy (ASYNCcnt-tracked), 16B per lane ----
__device__ __forceinline__ void async_load_b128(unsigned lds_off, const void* gaddr) {
  asm volatile("global_load_async_to_lds_b128 %0, %1, off"
               :: "v"(lds_off), "v"((unsigned long long)(uintptr_t)gaddr)
               : "memory");
}
__device__ __forceinline__ void wait_asynccnt0() {
  asm volatile("s_wait_asynccnt 0x0" ::: "memory");
}

// ---------------- fp32 -> bf16 conversions (one-time) ----------------
__global__ void convert_weights(const float* __restrict__ wih,
                                const float* __restrict__ whh,
                                unsigned short* __restrict__ wih_bf,
                                unsigned short* __restrict__ whh_bf, int n) {
  int i = blockIdx.x * blockDim.x + threadIdx.x;
  if (i < n) {
    wih_bf[i] = f2bf_bits(wih[i]);
    whh_bf[i] = f2bf_bits(whh[i]);
  }
}

__global__ void convert_x(const float* __restrict__ x,
                          unsigned int* __restrict__ xbf_packed, int n4) {
  int i = blockIdx.x * blockDim.x + threadIdx.x;   // one float4 per thread
  if (i < n4) {
    float4 v = ((const float4*)x)[i];
    xbf_packed[i * 2 + 0] =
        (unsigned)f2bf_bits(v.x) | ((unsigned)f2bf_bits(v.y) << 16);
    xbf_packed[i * 2 + 1] =
        (unsigned)f2bf_bits(v.z) | ((unsigned)f2bf_bits(v.w) << 16);
  }
}

// ---------------- banded top-k window selection ----------------
__global__ __launch_bounds__(256) void topk_kernel(const float* __restrict__ x,
                                                   int* __restrict__ ids) {
  const int lane  = threadIdx.x & 31;
  const int gwave = (blockIdx.x * blockDim.x + threadIdx.x) >> 5; // sequence id
  if (gwave >= NSEQ) return;
  const int t = gwave & (T_SEQ - 1);
  const float* xt = x + (size_t)gwave * DFEAT;

  float xv[16];
#pragma unroll
  for (int i = 0; i < 4; ++i) {
    float4 v = ((const float4*)xt)[lane * 4 + i];
    xv[i * 4 + 0] = v.x; xv[i * 4 + 1] = v.y;
    xv[i * 4 + 2] = v.z; xv[i * 4 + 3] = v.w;
  }

  float sc[23];
#pragma unroll
  for (int c = 0; c < 23; ++c) {
    int j = t + c - (WIN - 1);
    float s = -1e30f;
    if (j >= 0 && j < T_SEQ) {
      const float* xj = x + ((size_t)gwave - t + j) * DFEAT;
      float p = 0.f;
#pragma unroll
      for (int i = 0; i < 4; ++i) {
        float4 v = ((const float4*)xj)[lane * 4 + i];
        p = fmaf(xv[i * 4 + 0], v.x, p);
        p = fmaf(xv[i * 4 + 1], v.y, p);
        p = fmaf(xv[i * 4 + 2], v.z, p);
        p = fmaf(xv[i * 4 + 3], v.w, p);
      }
#pragma unroll
      for (int m = 16; m >= 1; m >>= 1) p += __shfl_xor(p, m, 32);
      s = p;
    }
    sc[c] = s;
  }

  // top-12 by value, ties -> smaller index (matches lax.top_k ordering)
  int chosen[WIN];
  unsigned used = 0;
#pragma unroll
  for (int k = 0; k < WIN; ++k) {
    float best = -1e38f;
    int bi = 0;
#pragma unroll
    for (int c = 0; c < 23; ++c) {
      bool ok = !((used >> c) & 1u) && (sc[c] > best);
      if (ok) { best = sc[c]; bi = c; }
    }
    used |= (1u << bi);
    chosen[k] = t + bi - (WIN - 1);
  }
  // sort ascending (constant-index bubble network)
#pragma unroll
  for (int i = 0; i < WIN - 1; ++i)
#pragma unroll
    for (int j = 0; j < WIN - 1 - i; ++j)
      if (chosen[j] > chosen[j + 1]) {
        int tmp = chosen[j]; chosen[j] = chosen[j + 1]; chosen[j + 1] = tmp;
      }
  if (lane == 0) {
#pragma unroll
    for (int k = 0; k < WIN; ++k) ids[gwave * WIN + k] = chosen[k];
  }
}

// ---------------- fused windowed GRU (bf16 WMMA, fp32 state) ----------------
__global__ __launch_bounds__(256, 1) void gru_kernel(
    const float* __restrict__ x,              // fp32, residual add
    const unsigned short* __restrict__ xbf,   // bf16 copy of x, gather source
    const unsigned short* __restrict__ wih,   // [1536][512] bf16
    const unsigned short* __restrict__ whh,   // [1536][512] bf16
    const float* __restrict__ bih,
    const float* __restrict__ bhh,
    const int* __restrict__ ids,
    float* __restrict__ out) {
  __shared__ __align__(16) unsigned short xg_lds[2][MSEQ][DFEAT]; // double-buffered
  __shared__ __align__(16) unsigned short h_lds[MSEQ][DFEAT];

  const int tid  = threadIdx.x;
  const int wave = tid >> 5;
  const int lane = tid & 31;
  const int hi   = lane >> 4;  // lane half
  const int lc   = lane & 15;
  const int seqbase = blockIdx.x * MSEQ;
  const int dbase   = wave * DSLICE;

  // h0 = 0
  for (int i = tid; i < MSEQ * DFEAT / 2; i += 256)
    ((unsigned int*)&h_lds[0][0])[i] = 0u;

  float hreg[MT][NSUB][8];
#pragma unroll
  for (int m = 0; m < MT; ++m)
#pragma unroll
    for (int s = 0; s < NSUB; ++s)
#pragma unroll
      for (int j = 0; j < 8; ++j) hreg[m][s][j] = 0.f;

  // per-lane gate biases for this wave's hidden slice
  float br[NSUB], bz[NSUB], bxn[NSUB], bhn[NSUB];
#pragma unroll
  for (int s = 0; s < NSUB; ++s) {
    int d  = dbase + s * 16 + lc;
    br[s]  = bih[d] + bhh[d];
    bz[s]  = bih[512 + d] + bhh[512 + d];
    bxn[s] = bih[1024 + d];
    bhn[s] = bhh[1024 + d];
  }

  // async gather of one step's 32 rows (bf16) into xg_lds[buf]:
  // 8 threads per row, 8 x b128 per thread, no VGPR round-trip.
  const int grow  = tid >> 3;          // 0..31
  const int gcol  = (tid & 7) * 64;    // element offset within row
  const int ggseq = seqbase + grow;
  const int gbb   = ggseq >> 11;
  auto issue_gather = [&](int step, int buf) {
    const int id = ids[ggseq * WIN + step];
    const unsigned short* src =
        xbf + ((size_t)(gbb << 11) + (size_t)id) * DFEAT + gcol;
    unsigned lds0 = (unsigned)(uintptr_t)&xg_lds[buf][grow][gcol];
#pragma unroll
    for (int i = 0; i < 8; ++i)
      async_load_b128(lds0 + i * 16, src + i * 8);
  };

  // prologue: prefetch step 0
  issue_gather(0, 0);
  wait_asynccnt0();
  __syncthreads();   // h_lds zeroed + step-0 gather visible

  const char* pih = (const char*)wih;
  const char* phh = (const char*)whh;

  for (int step = 0; step < WIN; ++step) {
    const int buf = step & 1;
    if (step + 1 < WIN) issue_gather(step + 1, buf ^ 1);  // overlap with GEMMs
    const unsigned short(*xg)[DFEAT] = xg_lds[buf];

    // ---- gate GEMMs + pointwise GRU update, all state in registers ----
#pragma unroll
    for (int sub = 0; sub < NSUB; ++sub) {
      const int dcol = dbase + sub * 16;  // r cols; z: 512+dcol; n: 1024+dcol
      f32x8 racc[MT], zacc[MT], xnacc[MT], hnacc[MT];
#pragma unroll
      for (int m = 0; m < MT; ++m) {
        racc[m] = {}; zacc[m] = {}; xnacc[m] = {}; hnacc[m] = {};
      }

      // per-lane 32-bit byte offset into [1536][512] bf16 weights:
      // col = dcol+lc (r gate), K chunk starts at hi*16; advances 64B per kt.
      // z/n gate sections are uniform displacements folded into the 24-bit
      // instruction offset, one voff += 64 per kt.
      unsigned voff = ((unsigned)(dcol + lc) * DFEAT + (unsigned)hi * 16) * 2u;

      for (int kt = 0; kt < KT; ++kt, voff += 64u) {
        // A fragments (16x32 bf16): lane = row lc, K chunks at koff, koff+8
        const int koff = kt * 32 + hi * 16;
        bf16x16 ax[MT], ah[MT];
#pragma unroll
        for (int m = 0; m < MT; ++m) {
          const int arow = m * 16 + lc;
          union { bf16x16 v; u32x4 u[2]; } fa, fh;
          fa.u[0] = *(const u32x4*)&xg[arow][koff];
          fa.u[1] = *(const u32x4*)&xg[arow][koff + 8];
          fh.u[0] = *(const u32x4*)&h_lds[arow][koff];
          fh.u[1] = *(const u32x4*)&h_lds[arow][koff + 8];
          ax[m] = fa.v;
          ah[m] = fh.v;
        }
        // B fragment (32x16 bf16): lane = col lc, 16 contiguous K (32 bytes)
        auto loadB = [&](const char* base, unsigned sect) {
          union { bf16x16 v; u32x4 u[2]; } f;
          f.u[0] = *(const u32x4*)(base + (size_t)voff + sect);
          f.u[1] = *(const u32x4*)(base + (size_t)voff + sect + 16u);
          return f.v;
        };
        bf16x16 b;
        b = loadB(pih, 0u);
#pragma unroll
        for (int m = 0; m < MT; ++m) racc[m] = wmma_bf16(ax[m], b, racc[m]);
        b = loadB(phh, 0u);
#pragma unroll
        for (int m = 0; m < MT; ++m) racc[m] = wmma_bf16(ah[m], b, racc[m]);
        b = loadB(pih, SECT_Z);
#pragma unroll
        for (int m = 0; m < MT; ++m) zacc[m] = wmma_bf16(ax[m], b, zacc[m]);
        b = loadB(phh, SECT_Z);
#pragma unroll
        for (int m = 0; m < MT; ++m) zacc[m] = wmma_bf16(ah[m], b, zacc[m]);
        b = loadB(pih, SECT_N);
#pragma unroll
        for (int m = 0; m < MT; ++m) xnacc[m] = wmma_bf16(ax[m], b, xnacc[m]);
        b = loadB(phh, SECT_N);
#pragma unroll
        for (int m = 0; m < MT; ++m) hnacc[m] = wmma_bf16(ah[m], b, hnacc[m]);
      }

      // pointwise GRU update; C element (j,lane) = (m=j+8*hi, n=lc)
#pragma unroll
      for (int m = 0; m < MT; ++m) {
#pragma unroll
        for (int j = 0; j < 8; ++j) {
          float r  = fast_sigmoid(racc[m][j] + br[sub]);
          float z  = fast_sigmoid(zacc[m][j] + bz[sub]);
          float nn = fast_tanh(xnacc[m][j] + bxn[sub] + r * (hnacc[m][j] + bhn[sub]));
          float hp = hreg[m][sub][j];
          hreg[m][sub][j] = (1.f - z) * nn + z * hp;
        }
      }
    }

    __syncthreads();  // all reads of h_lds / xg[buf] for this step done
    // publish new h (bf16) for next step's A operand
#pragma unroll
    for (int m = 0; m < MT; ++m)
#pragma unroll
      for (int s = 0; s < NSUB; ++s) {
        int d = dbase + s * 16 + lc;
#pragma unroll
        for (int j = 0; j < 8; ++j)
          h_lds[m * 16 + hi * 8 + j][d] = f2bf_bits(hreg[m][s][j]);
      }
    wait_asynccnt0();  // next step's gather buffer resident (own loads)
    __syncthreads();   // h writes + gathered buffer visible to all waves
  }

  // ---- residual add + store ----
#pragma unroll
  for (int m = 0; m < MT; ++m)
#pragma unroll
    for (int s = 0; s < NSUB; ++s) {
      int d = dbase + s * 16 + lc;
#pragma unroll
      for (int j = 0; j < 8; ++j) {
        int gseq = seqbase + m * 16 + hi * 8 + j;
        size_t off = (size_t)gseq * DFEAT + d;
        out[off] = hreg[m][s][j] + x[off];
      }
    }
}

extern "C" void kernel_launch(void* const* d_in, const int* in_sizes, int n_in,
                              void* d_out, int out_size, void* d_ws, size_t ws_size,
                              hipStream_t stream) {
  const float* x   = (const float*)d_in[0];
  const float* wih = (const float*)d_in[1];
  const float* whh = (const float*)d_in[2];
  const float* bih = (const float*)d_in[3];
  const float* bhh = (const float*)d_in[4];
  float* out = (float*)d_out;

  char* ws = (char*)d_ws;
  int* ids = (int*)ws;                                          // NSEQ*12 ints
  unsigned short* wih_bf =
      (unsigned short*)(ws + (size_t)NSEQ * WIN * sizeof(int));  // 16B aligned
  unsigned short* whh_bf = wih_bf + (size_t)GATE3 * DFEAT;
  unsigned short* xbf    = whh_bf + (size_t)GATE3 * DFEAT;       // 16B aligned

  const int nw = GATE3 * DFEAT;
  convert_weights<<<(nw + 255) / 256, 256, 0, stream>>>(wih, whh, wih_bf, whh_bf, nw);
  const int nx4 = NSEQ * DFEAT / 4;
  convert_x<<<(nx4 + 255) / 256, 256, 0, stream>>>(x, (unsigned int*)xbf, nx4);
  topk_kernel<<<(NSEQ * 32) / 256, 256, 0, stream>>>(x, ids);
  gru_kernel<<<NSEQ / MSEQ, 256, 0, stream>>>(x, xbf, wih_bf, whh_bf, bih, bhh,
                                              ids, out);
}
